// PVLSTM_50775103373420
// MI455X (gfx1250) — matrix operand order repaired and checked by reading, not compiled, not run on hardware
//
#include <hip/hip_runtime.h>

typedef __attribute__((ext_vector_type(16))) __bf16 v16bf;
typedef __attribute__((ext_vector_type(8)))  float  v8f;
typedef __attribute__((ext_vector_type(4)))  unsigned int u32x4;

#define B_  4096
#define T_  50
#define D_  96
#define H_  512
#define G_  2048
#define PF_ 25

__device__ __forceinline__ unsigned short f2bfu(float f) {
  unsigned u = __float_as_uint(f);
  u += 0x7FFFu + ((u >> 16) & 1u);
  return (unsigned short)(u >> 16);
}
__device__ __forceinline__ float bfu2f(unsigned short s) {
  return __uint_as_float(((unsigned)s) << 16);
}
__device__ __forceinline__ float sigmoidf_(float x) {
  return 1.0f / (1.0f + __expf(-x));
}

// CDNA5 async global->LDS copy (16B per lane), tracked by ASYNCcnt.
__device__ __forceinline__ void async_b128(unsigned lds_off, const void* gaddr) {
  asm volatile("global_load_async_to_lds_b128 %0, %1, off"
               :: "v"(lds_off), "v"(gaddr)
               : "memory");
}
__device__ __forceinline__ void wait_async0() {
  asm volatile("s_wait_asynccnt 0" ::: "memory");
}
__device__ __forceinline__ unsigned lds_off_of(const void* p) {
  return (unsigned)(size_t)p;   // low 32 bits of a generic LDS pointer = wave-relative LDS offset
}

// Build a per-lane v16bf fragment from two 16-byte LDS chunks.
__device__ __forceinline__ v16bf frag16(const unsigned short* p0, const unsigned short* p1) {
  union { v16bf v; u32x4 u[2]; } t;
  t.u[0] = *(const u32x4*)p0;
  t.u[1] = *(const u32x4*)p1;
  return t.v;
}
// A matrix 16x32 bf16 (ISA 7.12.2): lane<16 holds K=0..7,16..23 of row M=lane;
// lane>=16 holds K=8..15,24..31.  LDS row = 32 shorts.
__device__ __forceinline__ v16bf load_fragA(const unsigned short* sA, int row, int half) {
  const unsigned short* base = sA + row * 32;
  return frag16(base + half * 8, base + 16 + half * 8);
}
// B matrix 32x16 bf16: lane<16 holds K=0..7,16..23 of col N=lane; lane>=16 holds K=8..15,24..31.
__device__ __forceinline__ v16bf load_fragB(const unsigned short* sB, int nrow, int half) {
  const unsigned short* base = sB + nrow * 32;
  return frag16(base + half * 8, base + 16 + half * 8);
}
__device__ __forceinline__ v8f wmma_bf16(v16bf a, v16bf b, v8f c) {
  return __builtin_amdgcn_wmma_f32_16x16x32_bf16(false, a, false, b, (short)0, c, false, false);
}

// Fused LSTM step:  gates = A1 @ W1^T + Hprev @ W2^T + bias (gate-interleaved),
// then c' = sig(f)*c + sig(i)*tanh(g), h' = sig(o)*tanh(c').
// A1 is either f32 (optionally a difference A1f-A1fb, for velocity) or bf16 (A1h).
// Block tile 128(M) x 128(N); 8 wave32s in 2x4; wave tile 64x32 = 4x2 WMMA tiles.
// Double-buffered LDS staging fed by async global->LDS copies.
__global__ __launch_bounds__(256)
void lstm_step_kernel(const float* __restrict__ A1f, const float* __restrict__ A1fb,
                      const unsigned short* __restrict__ A1h, int lda1, int K1,
                      const unsigned short* __restrict__ W1,
                      const unsigned short* __restrict__ Hprev,
                      const unsigned short* __restrict__ W2,
                      const float* __restrict__ bias,
                      const float* __restrict__ c_in,
                      float* __restrict__ c_out,
                      unsigned short* __restrict__ h_out)
{
  __shared__ __align__(16) char smem[65536];
  // ping-pong staging: buf b -> A tile at b*16384, B tile at b*16384 + 8192
  float* sC = (float*)smem;   // 128x128 f32 epilogue reuse (64KB)

  const int tid   = threadIdx.x;
  const int lane  = tid & 31;
  const int wave  = tid >> 5;
  const int waveM = wave & 1;
  const int waveN = wave >> 1;
  const int half  = lane >> 4;
  const int l16   = lane & 15;

  const int m0 = blockIdx.x * 128;   // batch rows
  const int n0 = blockIdx.y * 128;   // interleaved gate columns

  v8f acc[4][2];
  const v8f vz = {0.f, 0.f, 0.f, 0.f, 0.f, 0.f, 0.f, 0.f};
#pragma unroll
  for (int i = 0; i < 4; ++i) { acc[i][0] = vz; acc[i][1] = vz; }

  const int ar = tid >> 1;          // tile row loaded by this thread
  const int ac = (tid & 1) << 4;    // 16-element K sub-chunk
  const int gm = m0 + ar;

  const int nk1 = K1 >> 5;          // phase-0 chunks
  const int nk  = nk1 + (H_ >> 5);  // total K chunks

  // Issue the loads for K-chunk ci into staging buffer `buf` (runtime GEPs only).
  auto issue = [&](int ci, int buf) {
    const int  p1  = (ci >= nk1);
    const int  k0  = (p1 ? (ci - nk1) : ci) << 5;
    const int  K   = p1 ? H_ : K1;
    const unsigned short* W = p1 ? W2 : W1;
    unsigned short* dA = (unsigned short*)(smem + buf * 16384) + ar * 32 + ac;
    unsigned short* dB = (unsigned short*)(smem + buf * 16384 + 8192) + ar * 32 + ac;
    // B tile (weights, always bf16): async direct to LDS
    {
      const unsigned short* src = W + (size_t)(n0 + ar) * K + k0 + ac;
      async_b128(lds_off_of(dB),     src);
      async_b128(lds_off_of(dB + 8), src + 8);
    }
    // A tile
    if (p1) {
      const unsigned short* src = Hprev + gm * H_ + k0 + ac;
      async_b128(lds_off_of(dA),     src);
      async_b128(lds_off_of(dA + 8), src + 8);
    } else if (A1h) {
      const unsigned short* src = A1h + gm * lda1 + k0 + ac;
      async_b128(lds_off_of(dA),     src);
      async_b128(lds_off_of(dA + 8), src + 8);
    } else if (A1fb) {
      const float* s0 = A1f  + gm * lda1 + k0 + ac;
      const float* s1 = A1fb + gm * lda1 + k0 + ac;
#pragma unroll
      for (int j = 0; j < 16; ++j) dA[j] = f2bfu(s0[j] - s1[j]);
    } else {
      const float* s0 = A1f + gm * lda1 + k0 + ac;
#pragma unroll
      for (int j = 0; j < 16; ++j) dA[j] = f2bfu(s0[j]);
    }
  };

  issue(0, 0);
#pragma unroll 1
  for (int ci = 0; ci < nk; ++ci) {
    const int buf = ci & 1;
    wait_async0();        // this wave's copies for chunk ci have landed in LDS
    __syncthreads();      // all waves' copies visible; prior reads of buf^1 retired
    if (ci + 1 < nk) issue(ci + 1, buf ^ 1);   // overlap next-chunk DMA with WMMA
    const unsigned short* sA = (const unsigned short*)(smem + buf * 16384);
    const unsigned short* sB = (const unsigned short*)(smem + buf * 16384 + 8192);
    v16bf b0 = load_fragB(sB, waveN * 32 + l16, half);
    v16bf b1 = load_fragB(sB, waveN * 32 + 16 + l16, half);
#pragma unroll
    for (int mi = 0; mi < 4; ++mi) {
      v16bf a = load_fragA(sA, waveM * 64 + mi * 16 + l16, half);
      acc[mi][0] = wmma_bf16(a, b0, acc[mi][0]);
      acc[mi][1] = wmma_bf16(a, b1, acc[mi][1]);
    }
  }

  // --- LSTM cell epilogue: regroup interleaved gates via LDS ---
  __syncthreads();
#pragma unroll
  for (int mi = 0; mi < 4; ++mi) {
#pragma unroll
    for (int ni = 0; ni < 2; ++ni) {
      const int col = waveN * 32 + ni * 16 + l16;
#pragma unroll
      for (int v = 0; v < 8; ++v) {
        const int row = waveM * 64 + mi * 16 + half * 8 + v;  // C layout: VGPR v -> M=v / M=8+v
        sC[row * 128 + col] = acc[mi][ni][v];
      }
    }
  }
  __syncthreads();

#pragma unroll
  for (int i = 0; i < 16; ++i) {
    const int p = tid * 16 + i;          // 128 rows x 32 hidden units
    const int row = p >> 5;
    const int hid = p & 31;
    const int col = hid << 2;
    const float gi = sC[row * 128 + col + 0] + bias[n0 + col + 0];
    const float gf = sC[row * 128 + col + 1] + bias[n0 + col + 1];
    const float gg = sC[row * 128 + col + 2] + bias[n0 + col + 2];
    const float go = sC[row * 128 + col + 3] + bias[n0 + col + 3];
    const int gb = m0 + row;
    const int gh = (n0 >> 2) + hid;
    const float cprev = c_in[gb * H_ + gh];
    const float iv = sigmoidf_(gi);
    const float fv = sigmoidf_(gf);
    const float gv = tanhf(gg);
    const float ov = sigmoidf_(go);
    const float cn = fv * cprev + iv * gv;
    c_out[gb * H_ + gh] = cn;
    h_out[gb * H_ + gh] = f2bfu(ov * tanhf(cn));
  }
}

// FC + hardtanh: out = clip(h1 @ fcW^T + b, ±1). M=4096, N=96, K=512.
// 8 waves, each owns 16 rows x full 96 cols (1x6 WMMA tiles).
__global__ __launch_bounds__(256)
void fc_step_kernel(const unsigned short* __restrict__ A,
                    const unsigned short* __restrict__ W,
                    const float* __restrict__ bias,
                    float* __restrict__ vel_out,       // pred_vel + t*D, row stride PF*D
                    unsigned short* __restrict__ x_next)
{
  __shared__ __align__(16) unsigned short sAB[128 * 32 + 96 * 32];
  unsigned short* sA = sAB;
  unsigned short* sB = sAB + 128 * 32;

  const int tid  = threadIdx.x;
  const int lane = tid & 31;
  const int wave = tid >> 5;
  const int half = lane >> 4;
  const int l16  = lane & 15;
  const int m0   = blockIdx.x * 128;

  v8f acc[6];
  const v8f vz = {0.f, 0.f, 0.f, 0.f, 0.f, 0.f, 0.f, 0.f};
#pragma unroll
  for (int j = 0; j < 6; ++j) acc[j] = vz;

  const int ar = tid >> 1;
  const int ac = (tid & 1) << 4;

#pragma unroll 1
  for (int k0 = 0; k0 < H_; k0 += 32) {
    __syncthreads();   // prior-iteration LDS reads retired
    {
      const unsigned short* src = A + (size_t)(m0 + ar) * H_ + k0 + ac;
      unsigned short* dst = sA + ar * 32 + ac;
      async_b128(lds_off_of(dst),     src);
      async_b128(lds_off_of(dst + 8), src + 8);
    }
    if (tid < 192) {
      const unsigned short* src = W + (size_t)ar * H_ + k0 + ac;
      unsigned short* dst = sB + ar * 32 + ac;
      async_b128(lds_off_of(dst),     src);
      async_b128(lds_off_of(dst + 8), src + 8);
    }
    wait_async0();
    __syncthreads();
    v16bf a = load_fragA(sA, wave * 16 + l16, half);
#pragma unroll
    for (int j = 0; j < 6; ++j) {
      v16bf b = load_fragB(sB, j * 16 + l16, half);
      acc[j] = wmma_bf16(a, b, acc[j]);
    }
  }

#pragma unroll
  for (int j = 0; j < 6; ++j) {
    const int col = j * 16 + l16;
    const float bv = bias[col];
#pragma unroll
    for (int v = 0; v < 8; ++v) {
      const int grow = m0 + wave * 16 + half * 8 + v;
      float val = acc[j][v] + bv;
      val = fminf(1.0f, fmaxf(-1.0f, val));
      vel_out[(size_t)grow * (PF_ * D_) + col] = val;
      x_next[(size_t)grow * D_ + col] = f2bfu(val);
    }
  }
}

// ---- small helper kernels ----
__global__ void pack_w_kernel(const float* __restrict__ W, unsigned short* __restrict__ out, int K) {
  int idx = blockIdx.x * blockDim.x + threadIdx.x;
  int n = G_ * K;
  if (idx >= n) return;
  int r = idx / K, k = idx - r * K;
  int oldr = (r & 3) * H_ + (r >> 2);   // interleave i,f,g,o per hidden unit
  out[idx] = f2bfu(W[(size_t)oldr * K + k]);
}
__global__ void pack_b_kernel(const float* bih, const float* bhh, float* out) {
  int r = blockIdx.x * blockDim.x + threadIdx.x;
  if (r >= G_) return;
  int oldr = (r & 3) * H_ + (r >> 2);
  out[r] = bih[oldr] + bhh[oldr];
}
__global__ void f32_to_bf16_kernel(const float* in, unsigned short* out, int n) {
  int i = blockIdx.x * blockDim.x + threadIdx.x;
  if (i < n) out[i] = f2bfu(in[i]);
}
__global__ void zero_h_kernel(unsigned short* p, int n) {
  int i = blockIdx.x * blockDim.x + threadIdx.x;
  if (i < n) p[i] = 0;
}
__global__ void zero_c_kernel(float* p, int n) {
  int i = blockIdx.x * blockDim.x + threadIdx.x;
  if (i < n) p[i] = 0.0f;
}
__global__ void add_h_kernel(unsigned short* io, const unsigned short* add, int n) {
  int i = blockIdx.x * blockDim.x + threadIdx.x;
  if (i < n) io[i] = f2bfu(bfu2f(io[i]) + bfu2f(add[i]));
}
__global__ void add_c_kernel(float* io, const float* add, int n) {
  int i = blockIdx.x * blockDim.x + threadIdx.x;
  if (i < n) io[i] += add[i];
}
__global__ void last_vel_kernel(const float* __restrict__ pose, unsigned short* __restrict__ out) {
  int idx = blockIdx.x * blockDim.x + threadIdx.x;
  if (idx >= B_ * D_) return;
  int b = idx / D_, d = idx - b * D_;
  const float* pr = pose + (size_t)b * (T_ * D_);
  out[idx] = f2bfu(pr[(T_ - 1) * D_ + d] - pr[(T_ - 2) * D_ + d]);
}
__global__ void cumsum_kernel(const float* __restrict__ vel, const float* __restrict__ pose,
                              float* __restrict__ pout) {
  int idx = blockIdx.x * blockDim.x + threadIdx.x;
  if (idx >= B_ * D_) return;
  int b = idx / D_, d = idx - b * D_;
  float accv = pose[(size_t)b * (T_ * D_) + (T_ - 1) * D_ + d];
  for (int t = 0; t < PF_; ++t) {
    accv += vel[(size_t)b * (PF_ * D_) + t * D_ + d];
    pout[(size_t)b * (PF_ * D_) + t * D_ + d] = accv;
  }
}

extern "C" void kernel_launch(void* const* d_in, const int* in_sizes, int n_in,
                              void* d_out, int out_size, void* d_ws, size_t ws_size,
                              hipStream_t stream)
{
  (void)in_sizes; (void)n_in; (void)out_size; (void)ws_size;
  const float* pose = (const float*)d_in[0];
  const float* fcW  = (const float*)d_in[25];
  const float* fcb  = (const float*)d_in[26];

  char* ws = (char*)d_ws;
  size_t off = 0;
  auto alloc = [&](size_t bytes) -> char* {
    char* p = ws + off;
    off = (off + bytes + 255) & ~(size_t)255;
    return p;
  };

  unsigned short *W10[3], *W20[3], *W11[3], *W21[3];
  float *b0[3], *b1[3];
  for (int n = 0; n < 3; ++n) {
    W10[n] = (unsigned short*)alloc((size_t)G_ * D_ * 2);
    W20[n] = (unsigned short*)alloc((size_t)G_ * H_ * 2);
    W11[n] = (unsigned short*)alloc((size_t)G_ * H_ * 2);
    W21[n] = (unsigned short*)alloc((size_t)G_ * H_ * 2);
    b0[n]  = (float*)alloc((size_t)G_ * 4);
    b1[n]  = (float*)alloc((size_t)G_ * 4);
  }
  unsigned short* fcWp = (unsigned short*)alloc((size_t)D_ * H_ * 2);
  unsigned short* hpp[2][2]; float* cpp[2][2];
  for (int l = 0; l < 2; ++l)
    for (int p = 0; p < 2; ++p) {
      hpp[l][p] = (unsigned short*)alloc((size_t)B_ * H_ * 2);
      cpp[l][p] = (float*)alloc((size_t)B_ * H_ * 4);
    }
  unsigned short* hvf[2]; float* cvf[2];
  for (int l = 0; l < 2; ++l) {
    hvf[l] = (unsigned short*)alloc((size_t)B_ * H_ * 2);
    cvf[l] = (float*)alloc((size_t)B_ * H_ * 4);
  }
  unsigned short* xdec[2];
  for (int p = 0; p < 2; ++p) xdec[p] = (unsigned short*)alloc((size_t)B_ * D_ * 2);

  float* pred_pose = (float*)d_out;
  float* pred_vel  = pred_pose + (size_t)B_ * PF_ * D_;

  auto g1 = [](int n) { return dim3((unsigned)((n + 255) / 256)); };
  const dim3 gG(B_ / 128, G_ / 128);
  const int nBH = B_ * H_;

  // 1) pack weights (gate-interleaved) to bf16
  for (int n = 0; n < 3; ++n) {
    int base = 1 + n * 8;
    pack_w_kernel<<<g1(G_ * D_), 256, 0, stream>>>((const float*)d_in[base + 0], W10[n], D_);
    pack_w_kernel<<<g1(G_ * H_), 256, 0, stream>>>((const float*)d_in[base + 1], W20[n], H_);
    pack_w_kernel<<<g1(G_ * H_), 256, 0, stream>>>((const float*)d_in[base + 4], W11[n], H_);
    pack_w_kernel<<<g1(G_ * H_), 256, 0, stream>>>((const float*)d_in[base + 5], W21[n], H_);
    pack_b_kernel<<<g1(G_), 256, 0, stream>>>((const float*)d_in[base + 2], (const float*)d_in[base + 3], b0[n]);
    pack_b_kernel<<<g1(G_), 256, 0, stream>>>((const float*)d_in[base + 6], (const float*)d_in[base + 7], b1[n]);
  }
  f32_to_bf16_kernel<<<g1(D_ * H_), 256, 0, stream>>>(fcW, fcWp, D_ * H_);

  // 2) velocity encoder: 49 steps, input = pose[t+1]-pose[t] converted on the fly
  for (int l = 0; l < 2; ++l) {
    zero_h_kernel<<<g1(nBH), 256, 0, stream>>>(hpp[l][0], nBH);
    zero_c_kernel<<<g1(nBH), 256, 0, stream>>>(cpp[l][0], nBH);
  }
  for (int t = 0; t < T_ - 1; ++t) {
    int pi = t & 1, po = (t + 1) & 1;
    lstm_step_kernel<<<gG, 256, 0, stream>>>(pose + (t + 1) * D_, pose + t * D_, nullptr, T_ * D_, D_,
        W10[1], hpp[0][pi], W20[1], b0[1], cpp[0][pi], cpp[0][po], hpp[0][po]);
    lstm_step_kernel<<<gG, 256, 0, stream>>>(nullptr, nullptr, hpp[0][po], H_, H_,
        W11[1], hpp[1][pi], W21[1], b1[1], cpp[1][pi], cpp[1][po], hpp[1][po]);
  }
  for (int l = 0; l < 2; ++l) {  // vel finals live in slot (T-1)&1 == 1
    (void)hipMemcpyAsync(hvf[l], hpp[l][1], (size_t)nBH * 2, hipMemcpyDeviceToDevice, stream);
    (void)hipMemcpyAsync(cvf[l], cpp[l][1], (size_t)nBH * 4, hipMemcpyDeviceToDevice, stream);
  }

  // 3) pose encoder: 50 steps, finals land in slot 0
  for (int l = 0; l < 2; ++l) {
    zero_h_kernel<<<g1(nBH), 256, 0, stream>>>(hpp[l][0], nBH);
    zero_c_kernel<<<g1(nBH), 256, 0, stream>>>(cpp[l][0], nBH);
  }
  for (int t = 0; t < T_; ++t) {
    int pi = t & 1, po = (t + 1) & 1;
    lstm_step_kernel<<<gG, 256, 0, stream>>>(pose + t * D_, nullptr, nullptr, T_ * D_, D_,
        W10[0], hpp[0][pi], W20[0], b0[0], cpp[0][pi], cpp[0][po], hpp[0][po]);
    lstm_step_kernel<<<gG, 256, 0, stream>>>(nullptr, nullptr, hpp[0][po], H_, H_,
        W11[0], hpp[1][pi], W21[0], b1[0], cpp[1][pi], cpp[1][po], hpp[1][po]);
  }
  // 4) decoder init: h = h_p + h_v, c = c_p + c_v (in slot 0)
  for (int l = 0; l < 2; ++l) {
    add_h_kernel<<<g1(nBH), 256, 0, stream>>>(hpp[l][0], hvf[l], nBH);
    add_c_kernel<<<g1(nBH), 256, 0, stream>>>(cpp[l][0], cvf[l], nBH);
  }

  // 5) decoder: 25 steps of 2-layer LSTM + FC(hardtanh)
  last_vel_kernel<<<g1(B_ * D_), 256, 0, stream>>>(pose, xdec[0]);
  for (int t = 0; t < PF_; ++t) {
    int pi = t & 1, po = (t + 1) & 1;
    lstm_step_kernel<<<gG, 256, 0, stream>>>(nullptr, nullptr, xdec[pi], D_, D_,
        W10[2], hpp[0][pi], W20[2], b0[2], cpp[0][pi], cpp[0][po], hpp[0][po]);
    lstm_step_kernel<<<gG, 256, 0, stream>>>(nullptr, nullptr, hpp[0][po], H_, H_,
        W11[2], hpp[1][pi], W21[2], b1[2], cpp[1][pi], cpp[1][po], hpp[1][po]);
    fc_step_kernel<<<dim3(B_ / 128), 256, 0, stream>>>(hpp[1][po], fcWp, fcb,
        pred_vel + t * D_, xdec[po]);
  }

  // 6) pred_pose = cumsum(pred_vel) + pose[:, -1, :]
  cumsum_kernel<<<g1(B_ * D_), 256, 0, stream>>>(pred_vel, pose, pred_pose);
}
